// TokenUOTBlock_65360812310690
// MI455X (gfx1250) — compile-verified, matrix operand order checked by reading, not run on hardware
//
#include <hip/hip_runtime.h>

// ---------------------------------------------------------------------------
// TokenUOTBlock for MI455X (gfx1250): bf16 WMMA GEMMs (TDM-staged LDS tiles)
// + wave32 log-domain Sinkhorn.  B=1, N=1024, D=512, H=16, DH=32, K_ITER=20.
// ---------------------------------------------------------------------------

typedef __attribute__((ext_vector_type(16))) __bf16 v16bf;
typedef __attribute__((ext_vector_type(8)))  float  v8f;
typedef unsigned int u32x4 __attribute__((ext_vector_type(4)));
typedef int          i32x4 __attribute__((ext_vector_type(4)));
typedef int          i32x8 __attribute__((ext_vector_type(8)));

union Frag { v16bf v; uint4 u[2]; };

#if __has_builtin(__builtin_amdgcn_tensor_load_to_lds)
#define HAVE_TDM 1
#else
#define HAVE_TDM 0
#endif

// Load a 16x32 bf16 WMMA A/B fragment from a row-major (row, K) tile.
// CDNA5 16-bit layout: lanes 0-15 hold row r, K {0..7,16..23};
// lanes 16-31 hold row r, K {8..15,24..31}.  Works on global or LDS tiles.
__device__ __forceinline__ v16bf load_frag(const __bf16* base, int ld) {
  const int lane = threadIdx.x & 31;
  const int r    = lane & 15;
  const int hf   = lane >> 4;
  const __bf16* p = base + (size_t)r * ld + hf * 8;
  Frag f;
  f.u[0] = *reinterpret_cast<const uint4*>(p);
  f.u[1] = *reinterpret_cast<const uint4*>(p + 16);
  return f.v;
}

__device__ __forceinline__ float wave_sum(float v) {
  for (int off = 16; off; off >>= 1) v += __shfl_xor(v, off, 32);
  return v;
}

__device__ __forceinline__ void lse_acc(float& m, float& s, float x) {
  if (x > m) { s = s * __expf(m - x) + 1.f; m = x; }
  else       { s += __expf(x - m); }
}

__device__ __forceinline__ float lse_wave_reduce(float m, float s) {
  for (int off = 16; off; off >>= 1) {
    float om = __shfl_xor(m, off, 32);
    float os = __shfl_xor(s, off, 32);
    float nm = fmaxf(m, om);
    float e1 = (m == nm)  ? 1.f : __expf(m - nm);
    float e2 = (om == nm) ? 1.f : __expf(om - nm);
    s = s * e1 + os * e2;
    m = nm;
  }
  return m + __logf(s);
}

#if HAVE_TDM
// Issue a TDM 2D tile load (rows x cols bf16, row stride in elements) into LDS.
// D# packing per CDNA5 ISA ch.8 (groups 0/1; groups 2/3 zero => 2D tensor).
// clang-23 toolchain: 6-arg builtin (g0, g1, g2, g3, extra, cpol).
__device__ __forceinline__ void tdm_load_2d(const __bf16* gsrc, unsigned lds_off,
                                            int rows, int cols, int stride_elems) {
  unsigned long long ga = (unsigned long long)(const void*)gsrc;
  u32x4 g0;
  g0[0] = 1u;                                        // count=1, user descriptor
  g0[1] = lds_off;                                   // LDS byte address
  g0[2] = (unsigned)(ga & 0xffffffffu);              // global_addr[31:0]
  g0[3] = (unsigned)((ga >> 32) & 0x01ffffffu)       // global_addr[56:32]
          | (2u << 30);                              // type = 2 ("image")
  unsigned td0 = (unsigned)stride_elems;             // tensor_dim0 (elements)
  unsigned td1 = (unsigned)rows;                     // tensor_dim1
  unsigned long long s0 = (unsigned long long)stride_elems; // dim0 stride
  i32x8 g1;
  g1[0] = (int)(1u << 16);                           // data_size=1 (2 bytes)
  g1[1] = (int)((td0 & 0xffffu) << 16);              // [63:48]=tensor_dim0 lo16
  g1[2] = (int)((td0 >> 16) | ((td1 & 0xffffu) << 16));
  g1[3] = (int)((td1 >> 16) | ((unsigned)cols << 16)); // [127:112]=tile_dim0
  g1[4] = (int)((unsigned)rows & 0xffffu);           // tile_dim1; tile_dim2=0
  g1[5] = (int)(s0 & 0xffffffffu);                   // dim0_stride[31:0]
  g1[6] = (int)((s0 >> 32) & 0xffffu);               // dim0_stride[47:32]
  g1[7] = 0;                                         // dim1_stride (unused 2D)
  i32x4 z4 = {0, 0, 0, 0};
  i32x8 z8 = {0, 0, 0, 0, 0, 0, 0, 0};
  __builtin_amdgcn_tensor_load_to_lds(g0, g1, z4, z4, z8, 0);
}
#endif

// ---------------------------------------------------------------------------
// Generic GEMM: C[M,Nout] (f32) = A[M,K] (bf16) x W[Nout,K]^T (bf16)
// Block = 256 threads = 8 waves; block tile 64x128; K-step 32.
// A (64x32) and W (128x32) tiles staged to LDS via the Tensor Data Mover
// (one descriptor pair issued by wave 0, TENSORcnt wait, then ds_load frags).
// ---------------------------------------------------------------------------
__global__ __launch_bounds__(256)
void gemm_bf16(const __bf16* __restrict__ A, const __bf16* __restrict__ W,
               float* __restrict__ C, int M, int Nout, int K) {
  __shared__ __bf16 ldsA[64][32];    // 4 KB
  __shared__ __bf16 ldsW[128][32];   // 8 KB
  const int wid  = threadIdx.x >> 5;
  const int lane = threadIdx.x & 31;
  const int r  = lane & 15, hf = lane >> 4;
  const int mB = blockIdx.x * 64;
  const int nB = blockIdx.y * 128;
  const int ml = (wid >> 1) * 16;    // local m-tile
  const int nl = (wid & 1) * 64;     // local n-strip
  v8f acc0 = {}, acc1 = {}, acc2 = {}, acc3 = {};
  for (int k = 0; k < K; k += 32) {
#if HAVE_TDM
    if (wid == 0) {
      tdm_load_2d(A + (size_t)mB * K + k, (unsigned)(size_t)&ldsA[0][0], 64, 32, K);
      tdm_load_2d(W + (size_t)nB * K + k, (unsigned)(size_t)&ldsW[0][0], 128, 32, K);
      __builtin_amdgcn_s_wait_tensorcnt(0);
    }
    __syncthreads();
#else
    for (int t = threadIdx.x; t < 64 * 32; t += 256)
      ldsA[t >> 5][t & 31] = A[(size_t)(mB + (t >> 5)) * K + k + (t & 31)];
    for (int t = threadIdx.x; t < 128 * 32; t += 256)
      ldsW[t >> 5][t & 31] = W[(size_t)(nB + (t >> 5)) * K + k + (t & 31)];
    __syncthreads();
#endif
    v16bf a  = load_frag(&ldsA[ml][0], 32);
    v16bf b0 = load_frag(&ldsW[nl +  0][0], 32);
    v16bf b1 = load_frag(&ldsW[nl + 16][0], 32);
    v16bf b2 = load_frag(&ldsW[nl + 32][0], 32);
    v16bf b3 = load_frag(&ldsW[nl + 48][0], 32);
    acc0 = __builtin_amdgcn_wmma_f32_16x16x32_bf16(false, a, false, b0, (short)0, acc0, false, false);
    acc1 = __builtin_amdgcn_wmma_f32_16x16x32_bf16(false, a, false, b1, (short)0, acc1, false, false);
    acc2 = __builtin_amdgcn_wmma_f32_16x16x32_bf16(false, a, false, b2, (short)0, acc2, false, false);
    acc3 = __builtin_amdgcn_wmma_f32_16x16x32_bf16(false, a, false, b3, (short)0, acc3, false, false);
    __syncthreads();
  }
  for (int v = 0; v < 8; ++v) {
    int row = mB + ml + v + hf * 8;
    float* cr = C + (size_t)row * Nout + r;
    cr[nB + nl +  0] = acc0[v];
    cr[nB + nl + 16] = acc1[v];
    cr[nB + nl + 32] = acc2[v];
    cr[nB + nl + 48] = acc3[v];
  }
}

// ---------------------------------------------------------------------------
// Elementwise helpers
// ---------------------------------------------------------------------------
__global__ void f2bf_kernel(const float* __restrict__ x, __bf16* __restrict__ y, int n) {
  int i = blockIdx.x * 256 + threadIdx.x;
  if (i < n) y[i] = (__bf16)x[i];
}

__global__ void swiglu_kernel(const float* __restrict__ a1, const float* __restrict__ a3,
                              __bf16* __restrict__ g, int n) {
  int i = blockIdx.x * 256 + threadIdx.x;
  if (i < n) {
    float a = a1[i];
    float s = a / (1.f + __expf(-a));
    g[i] = (__bf16)(s * a3[i]);
  }
}

__global__ void add_res_mask_kernel(const float* __restrict__ base, const float* __restrict__ delta,
                                    const float* __restrict__ mask, float* __restrict__ out,
                                    int n, int D) {
  int i = blockIdx.x * 256 + threadIdx.x;
  if (i < n) out[i] = base[i] + delta[i] * mask[i / D];
}

// LayerNorm over D=512 with affine params, bf16 output. One block per row.
__global__ __launch_bounds__(256)
void ln_kernel(const float* __restrict__ x, const float* __restrict__ w,
               const float* __restrict__ bb, __bf16* __restrict__ y) {
  const int row = blockIdx.x;
  const int t   = threadIdx.x;
  __shared__ float s1[256], s2[256];
  float a = 0.f, q = 0.f;
  for (int d = t; d < 512; d += 256) {
    float v = x[(size_t)row * 512 + d];
    a += v; q += v * v;
  }
  s1[t] = a; s2[t] = q; __syncthreads();
  for (int o = 128; o; o >>= 1) {
    if (t < o) { s1[t] += s1[t + o]; s2[t] += s2[t + o]; }
    __syncthreads();
  }
  const float m   = s1[0] * (1.f / 512.f);
  const float var = s2[0] * (1.f / 512.f) - m * m;
  const float rs  = rsqrtf(var + 1e-5f);
  for (int d = t; d < 512; d += 256) {
    float v = x[(size_t)row * 512 + d];
    y[(size_t)row * 512 + d] = (__bf16)((v - m) * rs * w[d] + bb[d]);
  }
}

// Per-head LN of Q,K (over DH=32, one wave per (h,n)); V transposed to (H,32,N);
// sigmoid(G) stored fp32 in (H,N,32).
__global__ __launch_bounds__(256)
void post_qkvg_kernel(const float* __restrict__ Qf, const float* __restrict__ Kf,
                      const float* __restrict__ Vf, const float* __restrict__ Gf,
                      __bf16* __restrict__ Qh, __bf16* __restrict__ Kh,
                      __bf16* __restrict__ Vt, float* __restrict__ sg) {
  const int w = blockIdx.x * 8 + (threadIdx.x >> 5);  // 0..16383
  const int h = w >> 10;
  const int n = w & 1023;
  const int d = threadIdx.x & 31;
  const size_t src = (size_t)n * 512 + h * 32 + d;
  {
    float q = Qf[src];
    float m = wave_sum(q) * (1.f / 32.f);
    float dq = q - m;
    float vv = wave_sum(dq * dq) * (1.f / 32.f);
    Qh[((size_t)h * 1024 + n) * 32 + d] = (__bf16)(dq * rsqrtf(vv + 1e-5f));
  }
  {
    float q = Kf[src];
    float m = wave_sum(q) * (1.f / 32.f);
    float dq = q - m;
    float vv = wave_sum(dq * dq) * (1.f / 32.f);
    Kh[((size_t)h * 1024 + n) * 32 + d] = (__bf16)(dq * rsqrtf(vv + 1e-5f));
  }
  Vt[((size_t)h * 32 + d) * 1024 + n] = (__bf16)Vf[src];
  sg[((size_t)h * 1024 + n) * 32 + d] = 1.f / (1.f + __expf(-Gf[src]));
}

// ---------------------------------------------------------------------------
// logits: S = QK^T/sqrt(32) via one WMMA (K=32), + pos bias, - distance gate,
// /eps, mask -> logK (f32, H,N,N).  Block = 128 threads = 4 waves.
// ---------------------------------------------------------------------------
__global__ __launch_bounds__(128)
void logits_kernel(const __bf16* __restrict__ Qh, const __bf16* __restrict__ Kh,
                   const int* __restrict__ pos_bins, const float* __restrict__ pos_weight,
                   const float* __restrict__ w_dist_logit, const float* __restrict__ x_res,
                   const float* __restrict__ mask, const float* __restrict__ eps,
                   float* __restrict__ logK) {
  const int h    = blockIdx.z;
  const int wid  = threadIdx.x >> 5;
  const int lane = threadIdx.x & 31;
  const int r  = lane & 15, hf = lane >> 4;
  const int i0 = blockIdx.x * 16;
  const int j0 = blockIdx.y * 64 + wid * 16;
  v16bf a = load_frag(Qh + ((size_t)h * 1024 + i0) * 32, 32);
  v16bf b = load_frag(Kh + ((size_t)h * 1024 + j0) * 32, 32);
  v8f c = {};
  c = __builtin_amdgcn_wmma_f32_16x16x32_bf16(false, a, false, b, (short)0, c, false, false);
  const float scale   = 0.17677669529663687f;  // 1/sqrt(32)
  const float wd      = 1.f / (1.f + __expf(-w_dist_logit[h]));
  const float inv_eps = 1.f / eps[h];
  const int   j  = j0 + r;
  const float xj0 = x_res[j * 3 + 0], xj1 = x_res[j * 3 + 1], xj2 = x_res[j * 3 + 2];
  const float mj = mask[j];
  for (int v = 0; v < 8; ++v) {
    const int i = i0 + v + hf * 8;
    float s = c[v] * scale;
    int bin = pos_bins[(size_t)i * 1024 + j];
    float bias = pos_weight[bin * 16 + h];
    float d0 = x_res[i * 3 + 0] - xj0;
    float d1 = x_res[i * 3 + 1] - xj1;
    float d2 = x_res[i * 3 + 2] - xj2;
    float dist2 = d0 * d0 + d1 * d1 + d2 * d2;
    float logit = s + bias - wd * dist2 * 0.01f;
    float pm = mask[i] * mj;
    logK[((size_t)h * 1024 + i) * 1024 + j] = (pm > 0.f) ? logit * inv_eps : -1e9f;
  }
}

// ---------------------------------------------------------------------------
// Sinkhorn halves. logK (64MB) is L2-resident on MI455X (192MB L2).
// ---------------------------------------------------------------------------
__global__ __launch_bounds__(256)
void sink_row_kernel(const float* __restrict__ logK, const float* __restrict__ log_v,
                     const float* __restrict__ mu, const float* __restrict__ eps,
                     float* __restrict__ log_u) {
  const int row  = blockIdx.x * 8 + (threadIdx.x >> 5);   // (h,i) flat
  const int h    = row >> 10;
  const int lane = threadIdx.x & 31;
  const float* pk = logK + (size_t)row * 1024;
  const float* lv = log_v + h * 1024;
  float m = -INFINITY, s = 0.f;
  for (int j = lane; j < 1024; j += 32) lse_acc(m, s, pk[j] + lv[j]);
  float lse = lse_wave_reduce(m, s);
  if (lane == 0) {
    float fi = 1.f / (1.f + eps[h]);
    log_u[row] = fi * (__logf(fmaxf(mu[row], 1e-8f)) - lse);
  }
}

// Column pass, phase 1: per-column online-LSE over a 256-row chunk.
// Grid (4 j-tiles, 16 heads, 4 chunks) -> 256 blocks.
__global__ __launch_bounds__(256)
void sink_col_partial(const float* __restrict__ logK, const float* __restrict__ log_u,
                      float* __restrict__ pm, float* __restrict__ ps) {
  const int h     = blockIdx.y;
  const int chunk = blockIdx.z;
  const int j     = blockIdx.x * 256 + threadIdx.x;
  const float* pk = logK + (size_t)h * 1024 * 1024 + j;
  const float* lu = log_u + h * 1024;
  float m = -INFINITY, s = 0.f;
  const int i0 = chunk * 256;
  for (int i = i0; i < i0 + 256; ++i) lse_acc(m, s, pk[(size_t)i * 1024] + lu[i]);
  const size_t idx = ((size_t)chunk * 16 + h) * 1024 + j;
  pm[idx] = m;
  ps[idx] = s;
}

// Column pass, phase 2: combine 4 chunk partials -> log_v.
__global__ __launch_bounds__(256)
void sink_col_combine(const float* __restrict__ pm, const float* __restrict__ ps,
                      const float* __restrict__ nu, const float* __restrict__ eps,
                      float* __restrict__ log_v) {
  const int idx = blockIdx.x * 256 + threadIdx.x;   // (h,j) flat, 0..16383
  const int h = idx >> 10;
  const int j = idx & 1023;
  float m = -INFINITY, s = 0.f;
  for (int c = 0; c < 4; ++c) {
    const size_t p = ((size_t)c * 16 + h) * 1024 + j;
    float cm = pm[p], cs = ps[p];
    float nm = fmaxf(m, cm);
    float e1 = (m == nm)  ? 1.f : __expf(m - nm);
    float e2 = (cm == nm) ? 1.f : __expf(cm - nm);
    s = s * e1 + cs * e2;
    m = nm;
  }
  float lse = m + __logf(s);
  float fi = 1.f / (1.f + eps[h]);
  log_v[idx] = fi * (__logf(fmaxf(nu[idx], 1e-8f)) - lse);
}

// P = exp(logK + log_u_i + log_v_j) * mask, bf16; row sums fp32.
__global__ __launch_bounds__(256)
void p_kernel(const float* __restrict__ logK, const float* __restrict__ log_u,
              const float* __restrict__ log_v, const float* __restrict__ mask,
              __bf16* __restrict__ P, float* __restrict__ rowsum) {
  const int row  = blockIdx.x * 8 + (threadIdx.x >> 5);
  const int h    = row >> 10;
  const int i    = row & 1023;
  const int lane = threadIdx.x & 31;
  const float lu = log_u[row];
  const float mi = mask[i];
  const float* pk = logK + (size_t)row * 1024;
  __bf16* pp = P + (size_t)row * 1024;
  float acc = 0.f;
  for (int j = lane; j < 1024; j += 32) {
    float p = __expf(pk[j] + lu + log_v[h * 1024 + j]) * mi * mask[j];
    pp[j] = (__bf16)p;
    acc += p;
  }
  acc = wave_sum(acc);
  if (lane == 0) rowsum[row] = acc;
}

// ---------------------------------------------------------------------------
// attn = (P @ V) / rowsum * sigmoid(G), WMMA over j (K=1024 in 32-steps).
// ---------------------------------------------------------------------------
__global__ __launch_bounds__(256)
void attn_kernel(const __bf16* __restrict__ P, const __bf16* __restrict__ Vt,
                 const float* __restrict__ rowsum, const float* __restrict__ sg,
                 __bf16* __restrict__ o_bf) {
  const int h    = blockIdx.y;
  const int wid  = threadIdx.x >> 5;
  const int lane = threadIdx.x & 31;
  const int r  = lane & 15, hf = lane >> 4;
  const int i0 = blockIdx.x * 64 + (wid >> 1) * 16;
  const int d0 = (wid & 1) * 16;
  const __bf16* Abase = P  + ((size_t)h * 1024 + i0) * 1024;
  const __bf16* Bbase = Vt + ((size_t)h * 32 + d0) * 1024;
  v8f acc = {};
  for (int j = 0; j < 1024; j += 32) {
    v16bf a = load_frag(Abase + j, 1024);
    v16bf b = load_frag(Bbase + j, 1024);
    acc = __builtin_amdgcn_wmma_f32_16x16x32_bf16(false, a, false, b, (short)0, acc, false, false);
  }
  const int d = d0 + r;
  for (int v = 0; v < 8; ++v) {
    int i = i0 + v + hf * 8;
    float inv = 1.f / fmaxf(rowsum[h * 1024 + i], 1e-8f);
    float val = acc[v] * inv * sg[((size_t)h * 1024 + i) * 32 + d];
    o_bf[(size_t)i * 512 + h * 32 + d] = (__bf16)val;
  }
}

// ---------------------------------------------------------------------------
// Centroid EGNN coord update: x_out = x + mask*(G*x - sum_h g_h * Pn_h @ x)
// ---------------------------------------------------------------------------
__global__ __launch_bounds__(256)
void xcent_kernel(const __bf16* __restrict__ P, const float* __restrict__ rowsum,
                  const float* __restrict__ x_res, const float* __restrict__ gamma,
                  const float* __restrict__ mask, float* __restrict__ x_out) {
  const int i = blockIdx.x;
  const int t = threadIdx.x;
  __shared__ float r0[256], r1[256], r2[256];
  float a0 = 0.f, a1 = 0.f, a2 = 0.f;
  for (int h = 0; h < 16; ++h) {
    float g   = tanhf(gamma[h]) * (1.f / 16.f);
    float inv = g / fmaxf(rowsum[h * 1024 + i], 1e-8f);
    const __bf16* pr = P + ((size_t)h * 1024 + i) * 1024;
    float s0 = 0.f, s1 = 0.f, s2 = 0.f;
    for (int j = t; j < 1024; j += 256) {
      float p = (float)pr[j];
      s0 += p * x_res[j * 3 + 0];
      s1 += p * x_res[j * 3 + 1];
      s2 += p * x_res[j * 3 + 2];
    }
    a0 += inv * s0; a1 += inv * s1; a2 += inv * s2;
  }
  r0[t] = a0; r1[t] = a1; r2[t] = a2; __syncthreads();
  for (int o = 128; o; o >>= 1) {
    if (t < o) { r0[t] += r0[t + o]; r1[t] += r1[t + o]; r2[t] += r2[t + o]; }
    __syncthreads();
  }
  if (t == 0) {
    float G = 0.f;
    for (int h = 0; h < 16; ++h) G += tanhf(gamma[h]) * (1.f / 16.f);
    float mi = mask[i];
    x_out[i * 3 + 0] = x_res[i * 3 + 0] + mi * (G * x_res[i * 3 + 0] - r0[0]);
    x_out[i * 3 + 1] = x_res[i * 3 + 1] + mi * (G * x_res[i * 3 + 1] - r1[0]);
    x_out[i * 3 + 2] = x_res[i * 3 + 2] + mi * (G * x_res[i * 3 + 2] - r2[0]);
  }
}

// ---------------------------------------------------------------------------
// Host orchestration
// ---------------------------------------------------------------------------
extern "C" void kernel_launch(void* const* d_in, const int* in_sizes, int n_in,
                              void* d_out, int out_size, void* d_ws, size_t ws_size,
                              hipStream_t stream) {
  (void)in_sizes; (void)n_in; (void)out_size; (void)ws_size;
  const float* h_in    = (const float*)d_in[0];
  const float* x_res   = (const float*)d_in[1];
  const float* mu      = (const float*)d_in[2];
  const float* nu      = (const float*)d_in[3];
  const float* lu_prev = (const float*)d_in[4];
  const float* lv_prev = (const float*)d_in[5];
  const float* mask    = (const float*)d_in[6];
  const float* lnaw    = (const float*)d_in[7];
  const float* lnab    = (const float*)d_in[8];
  const float* wq      = (const float*)d_in[9];
  const float* wk      = (const float*)d_in[10];
  const float* wv      = (const float*)d_in[11];
  const float* wg      = (const float*)d_in[12];
  const float* wo      = (const float*)d_in[13];
  const float* gamma   = (const float*)d_in[14];
  const float* posw    = (const float*)d_in[15];
  const float* wdl     = (const float*)d_in[16];
  const float* lnfw    = (const float*)d_in[17];
  const float* lnfb    = (const float*)d_in[18];
  const float* sw1     = (const float*)d_in[19];
  const float* sw3     = (const float*)d_in[20];
  const float* sw2     = (const float*)d_in[21];
  const int*   pbins   = (const int*)d_in[22];
  const float* eps     = (const float*)d_in[23];

  char* wsb = (char*)d_ws;
  size_t off = 0;
  auto take = [&](size_t bytes) -> void* {
    void* p = wsb + off;
    off += (bytes + 255) & ~(size_t)255;
    return p;
  };
  float*  logK  = (float*)take(16UL * 1024 * 1024 * 4);   // 64 MB
  __bf16* P     = (__bf16*)take(16UL * 1024 * 1024 * 2);  // 32 MB
  __bf16* wq_b  = (__bf16*)take(512UL * 512 * 2);
  __bf16* wk_b  = (__bf16*)take(512UL * 512 * 2);
  __bf16* wv_b  = (__bf16*)take(512UL * 512 * 2);
  __bf16* wg_b  = (__bf16*)take(512UL * 512 * 2);
  __bf16* wo_b  = (__bf16*)take(512UL * 512 * 2);
  __bf16* w1_b  = (__bf16*)take(2048UL * 512 * 2);
  __bf16* w3_b  = (__bf16*)take(2048UL * 512 * 2);
  __bf16* w2_b  = (__bf16*)take(512UL * 2048 * 2);
  __bf16* hn    = (__bf16*)take(1024UL * 512 * 2);
  float*  Qf    = (float*)take(1024UL * 512 * 4);
  float*  Kf    = (float*)take(1024UL * 512 * 4);
  float*  Vf    = (float*)take(1024UL * 512 * 4);
  float*  Gf    = (float*)take(1024UL * 512 * 4);
  __bf16* Qh    = (__bf16*)take(16UL * 1024 * 32 * 2);
  __bf16* Kh    = (__bf16*)take(16UL * 1024 * 32 * 2);
  __bf16* Vt    = (__bf16*)take(16UL * 32 * 1024 * 2);
  float*  sg    = (float*)take(16UL * 1024 * 32 * 4);
  float*  lu    = (float*)take(16UL * 1024 * 4);
  float*  lv    = (float*)take(16UL * 1024 * 4);
  float*  rowsum= (float*)take(16UL * 1024 * 4);
  float*  pmb   = (float*)take(4UL * 16 * 1024 * 4);      // col-pass partial m
  float*  psb   = (float*)take(4UL * 16 * 1024 * 4);      // col-pass partial s
  __bf16* o_b   = (__bf16*)take(1024UL * 512 * 2);
  float*  tmpC  = (float*)take(1024UL * 512 * 4);
  float*  h_new = (float*)take(1024UL * 512 * 4);
  __bf16* h2    = (__bf16*)take(1024UL * 512 * 2);
  // FFN intermediates reuse the (then-dead) logK region
  float*  a1    = logK;                                // 1024x2048 f32
  float*  a3    = logK + 2UL * 1024 * 1024;            // 1024x2048 f32
  __bf16* gmid  = (__bf16*)(logK + 4UL * 1024 * 1024); // 1024x2048 bf16

  float* out_h  = (float*)d_out;            // 1024*512
  float* out_x  = out_h + 524288;           // 1024*3
  float* out_lu = out_h + 527360;           // 16*1024
  float* out_lv = out_h + 543744;           // 16*1024

  // 1) weights -> bf16
  f2bf_kernel<<<1024, 256, 0, stream>>>(wq, wq_b, 262144);
  f2bf_kernel<<<1024, 256, 0, stream>>>(wk, wk_b, 262144);
  f2bf_kernel<<<1024, 256, 0, stream>>>(wv, wv_b, 262144);
  f2bf_kernel<<<1024, 256, 0, stream>>>(wg, wg_b, 262144);
  f2bf_kernel<<<1024, 256, 0, stream>>>(wo, wo_b, 262144);
  f2bf_kernel<<<4096, 256, 0, stream>>>(sw1, w1_b, 1048576);
  f2bf_kernel<<<4096, 256, 0, stream>>>(sw3, w3_b, 1048576);
  f2bf_kernel<<<4096, 256, 0, stream>>>(sw2, w2_b, 1048576);

  // 2) LN(h) -> bf16
  ln_kernel<<<1024, 256, 0, stream>>>(h_in, lnaw, lnab, hn);

  // 3) QKVG projections (TDM-staged WMMA)
  dim3 g512(16, 4);
  gemm_bf16<<<g512, 256, 0, stream>>>(hn, wq_b, Qf, 1024, 512, 512);
  gemm_bf16<<<g512, 256, 0, stream>>>(hn, wk_b, Kf, 1024, 512, 512);
  gemm_bf16<<<g512, 256, 0, stream>>>(hn, wv_b, Vf, 1024, 512, 512);
  gemm_bf16<<<g512, 256, 0, stream>>>(hn, wg_b, Gf, 1024, 512, 512);
  post_qkvg_kernel<<<2048, 256, 0, stream>>>(Qf, Kf, Vf, Gf, Qh, Kh, Vt, sg);

  // 4) logits -> logK
  logits_kernel<<<dim3(64, 16, 16), 128, 0, stream>>>(Qh, Kh, pbins, posw, wdl,
                                                      x_res, mask, eps, logK);

  // 5) warm-started log-domain Sinkhorn, 20 iterations (logK L2-resident)
  (void)hipMemcpyAsync(lu, lu_prev, 16384 * 4, hipMemcpyDeviceToDevice, stream);
  (void)hipMemcpyAsync(lv, lv_prev, 16384 * 4, hipMemcpyDeviceToDevice, stream);
  for (int it = 0; it < 20; ++it) {
    sink_row_kernel<<<2048, 256, 0, stream>>>(logK, lv, mu, eps, lu);
    sink_col_partial<<<dim3(4, 16, 4), 256, 0, stream>>>(logK, lu, pmb, psb);
    sink_col_combine<<<64, 256, 0, stream>>>(pmb, psb, nu, eps, lv);
  }

  // 6) P (bf16) + row sums
  p_kernel<<<2048, 256, 0, stream>>>(logK, lu, lv, mask, P, rowsum);

  // 7) attn + output projection + residual
  attn_kernel<<<dim3(16, 16), 256, 0, stream>>>(P, Vt, rowsum, sg, o_b);
  gemm_bf16<<<g512, 256, 0, stream>>>(o_b, wo_b, tmpC, 1024, 512, 512);
  add_res_mask_kernel<<<2048, 256, 0, stream>>>(h_in, tmpC, mask, h_new, 524288, 512);

  // 8) centroid coordinate update -> out_x
  xcent_kernel<<<1024, 256, 0, stream>>>(P, rowsum, x_res, gamma, mask, out_x);

  // 9) SwiGLU FFN + residual -> out_h
  ln_kernel<<<1024, 256, 0, stream>>>(h_new, lnfw, lnfb, h2);
  dim3 g2048(16, 16);
  gemm_bf16<<<g2048, 256, 0, stream>>>(h2, w1_b, a1, 1024, 2048, 512);
  gemm_bf16<<<g2048, 256, 0, stream>>>(h2, w3_b, a3, 1024, 2048, 512);
  swiglu_kernel<<<8192, 256, 0, stream>>>(a1, a3, gmid, 1024 * 2048);
  gemm_bf16<<<g512, 256, 0, stream>>>(gmid, w2_b, tmpC, 1024, 512, 2048);
  add_res_mask_kernel<<<2048, 256, 0, stream>>>(h_new, tmpC, mask, out_h, 524288, 512);

  // 10) log_u / log_v outputs
  (void)hipMemcpyAsync(out_lu, lu, 16384 * 4, hipMemcpyDeviceToDevice, stream);
  (void)hipMemcpyAsync(out_lv, lv, 16384 * 4, hipMemcpyDeviceToDevice, stream);
}
